// ImprovedInductiveBuNNLayer_51049981280282
// MI455X (gfx1250) — matrix operand order repaired and checked by reading, not compiled, not run on hardware
//
#include <hip/hip_runtime.h>

typedef __attribute__((ext_vector_type(16))) _Float16 v16h;
typedef __attribute__((ext_vector_type(8)))  float    v8f;

#define TD    128   // total_dim = bundle pairs * 2
#define DIN   128
#define POSD  8
#define TIN   136   // DIN + POSD
#define NBUN  64
#define MAXDEG 8
#define TDIFF 1.0f

// ---------------- utility kernels ----------------

__global__ void zero_f32(float* p, int n) {
  int t = blockIdx.x * blockDim.x + threadIdx.x;
  if (t < n) p[t] = 0.0f;
}

__global__ void concat_kernel(const float* __restrict__ x, const float* __restrict__ pe,
                              float* __restrict__ xc, int N) {
  int t = blockIdx.x * blockDim.x + threadIdx.x;
  if (t >= N * TIN) return;
  int i = t / TIN, c = t % TIN;
  xc[t] = (c < DIN) ? x[(long)i * DIN + c] : pe[(long)i * POSD + (c - DIN)];
}

__global__ void count_kernel(const int* __restrict__ ei, float* __restrict__ cnt_tgt,
                             float* __restrict__ deg_src, int E) {
  int e = blockIdx.x * blockDim.x + threadIdx.x;
  if (e >= E) return;
  atomicAdd(&cnt_tgt[ei[E + e]], 1.0f);   // counts by tgt (mean aggr)
  atomicAdd(&deg_src[ei[e]], 1.0f);       // degree by src (RW Laplacian)
}

__global__ void dinv_kernel(const float* __restrict__ deg, float* __restrict__ dinv, int N) {
  int i = blockIdx.x * blockDim.x + threadIdx.x;
  if (i < N) dinv[i] = (deg[i] > 0.0f) ? 1.0f / deg[i] : 0.0f;
}

// agg[tgt[e], :] += feat[src[e], :], 4 channels per thread
__global__ void scatter_add_kernel(const int* __restrict__ ei, const float* __restrict__ feat,
                                   float* __restrict__ agg, int E, int F, int chunks) {
  int t = blockIdx.x * blockDim.x + threadIdx.x;
  if (t >= E * chunks) return;
  int e = t / chunks, c0 = (t % chunks) * 4;
  int s = ei[e], d = ei[E + e];
  const float* f = feat + (long)s * F + c0;
  float* o = agg + (long)d * F + c0;
  #pragma unroll
  for (int j = 0; j < 4; ++j) atomicAdd(o + j, f[j]);
}

// out = relu( (agg/max(cnt,1)) @ Wl^T + bl + xin @ Wr^T )
__global__ void sage_kernel(const float* __restrict__ xin, const float* __restrict__ agg,
                            const float* __restrict__ cnt,
                            const float* __restrict__ Wl, const float* __restrict__ bl,
                            const float* __restrict__ Wr,
                            float* __restrict__ out, int N, int Fin, int Fout) {
  int t = blockIdx.x * blockDim.x + threadIdx.x;
  if (t >= N * Fout) return;
  int i = t / Fout, c = t % Fout;
  float inv = 1.0f / fmaxf(cnt[i], 1.0f);
  const float* xr = xin + (long)i * Fin;
  const float* ar = agg + (long)i * Fin;
  const float* wl = Wl + (long)c * Fin;
  const float* wr = Wr + (long)c * Fin;
  float acc = bl[c];
  for (int k = 0; k < Fin; ++k)
    acc = fmaf(ar[k] * inv, wl[k], fmaf(xr[k], wr[k], acc));
  out[t] = fmaxf(acc, 0.0f);
}

// angles -> (cos, sin) per node
__global__ void angle_kernel(const float* __restrict__ h2, const float* __restrict__ Wa,
                             const float* __restrict__ ba, float* __restrict__ cs,
                             int N, int Fin) {
  int i = blockIdx.x * blockDim.x + threadIdx.x;
  if (i >= N) return;
  float a = ba[0];
  const float* hr = h2 + (long)i * Fin;
  for (int k = 0; k < Fin; ++k) a = fmaf(hr[k], Wa[k], a);
  cs[2 * i]     = __cosf(a);
  cs[2 * i + 1] = __sinf(a);
}

// apply per-node 2x2 rotation blockwise (transpose=0: O, transpose=1: O^T),
// optional bias add and relu
__global__ void rotate_kernel(const float* __restrict__ cs, const float* __restrict__ in,
                              const float* __restrict__ bias, float* __restrict__ out,
                              int N, int transpose, int relu) {
  int t = blockIdx.x * blockDim.x + threadIdx.x;
  if (t >= N * NBUN) return;
  int i = t >> 6, b = t & 63;
  float c = cs[2 * i], s = cs[2 * i + 1];
  float u0 = in[(long)i * TD + 2 * b];
  float u1 = in[(long)i * TD + 2 * b + 1];
  float o0, o1;
  if (!transpose) { o0 = c * u0 - s * u1; o1 = s * u0 + c * u1; }
  else            { o0 = c * u0 + s * u1; o1 = -s * u0 + c * u1; }
  if (bias) { o0 += bias[2 * b]; o1 += bias[2 * b + 1]; }
  if (relu) { o0 = fmaxf(o0, 0.0f); o1 = fmaxf(o1, 0.0f); }
  out[(long)i * TD + 2 * b]     = o0;
  out[(long)i * TD + 2 * b + 1] = o1;
}

// result = term = hsync (Taylor k=0 init)
__global__ void diff_init_kernel(const float* __restrict__ hsync, float* __restrict__ term,
                                 float* __restrict__ result, int n) {
  int t = blockIdx.x * blockDim.x + threadIdx.x;
  if (t >= n) return;
  float v = hsync[t];
  term[t] = v;
  result[t] = v;
}

// tmp[src,:] += dinv[src] * term[tgt,:]   (== (D^{-1}A) @ term)
__global__ void edge_apply_kernel(const int* __restrict__ ei, const float* __restrict__ dinv,
                                  const float* __restrict__ term, float* __restrict__ tmp,
                                  int E) {
  int t = blockIdx.x * blockDim.x + threadIdx.x;
  if (t >= E * 32) return;
  int e = t >> 5, q = t & 31;
  int s = ei[e], d = ei[E + e];
  float w = dinv[s];
  const float4 v = *(const float4*)(term + (long)d * TD + q * 4);
  float* o = tmp + (long)s * TD + q * 4;
  atomicAdd(o + 0, w * v.x);
  atomicAdd(o + 1, w * v.y);
  atomicAdd(o + 2, w * v.z);
  atomicAdd(o + 3, w * v.w);
}

// term = coef*(term - tmp);  result += term     (coef = -t/k)
__global__ void diff_update_kernel(float* __restrict__ term, const float* __restrict__ tmp,
                                   float* __restrict__ result, float coef, int n) {
  int t = blockIdx.x * blockDim.x + threadIdx.x;
  if (t >= n) return;
  float nt = coef * (term[t] - tmp[t]);
  term[t] = nt;
  result[t] += nt;
}

// ---------------- WMMA GEMM: Out[N x 128] = In[N x 128] @ W^T, W is 128x128 f32 ----------------
// One wave per 16x16 output tile, K=128 as 4 x v_wmma_f32_16x16x32_f16.
// W (as f16) and the 16-row A band are staged in LDS; fragments follow the CDNA5
// wave32 layouts: A 16x32 f16 (lanes 0-15 M, VGPR pairs hold K), B 32x16 f16
// (lane = column N, lanes 16-31 hold K+8..15, K+24..31), C/D 8 VGPRs (M = g*8 + r).
__global__ void __launch_bounds__(256) gemm_wmma_kernel(const float* __restrict__ In,
                                                        const float* __restrict__ W,
                                                        float* __restrict__ Out, int N) {
  __shared__ _Float16 sW[TD * TD];   // 32 KB
  __shared__ _Float16 sA[16 * TD];   // 4 KB
  int tid = threadIdx.x;
  int i0 = blockIdx.x * 16;
  for (int idx = tid; idx < TD * TD; idx += 256) sW[idx] = (_Float16)W[idx];
  for (int idx = tid; idx < 16 * TD; idx += 256)
    sA[idx] = (_Float16)In[(long)(i0 + (idx >> 7)) * TD + (idx & 127)];
  __syncthreads();

  int wave = tid >> 5;          // 8 waves -> 8 column tiles (128 out channels)
  int lane = tid & 31;
  int n0 = wave * 16;
  int l = lane & 15;            // A: row M within tile / B: column N within tile
  int g = lane >> 4;            // half-wave group
  v8f acc = {};
  #pragma unroll
  for (int k0 = 0; k0 < TD; k0 += 32) {
    v16h a, b;
    #pragma unroll
    for (int h = 0; h < 16; ++h) {
      int ka = (h < 8) ? (g * 8 + h) : (16 + g * 8 + (h - 8));
      a[h] = sA[l * TD + k0 + ka];                    // A[l, k0+ka]
      b[h] = sW[(n0 + l) * TD + k0 + ka];     // B[k, n] = W[n0+l, k]
    }
    acc = __builtin_amdgcn_wmma_f32_16x16x32_f16(
        /*neg_a=*/false, a, /*neg_b=*/false, b,
        /*c_mod=*/(short)0, acc, /*reuse_a=*/false, /*reuse_b=*/false);
  }
  #pragma unroll
  for (int r = 0; r < 8; ++r)
    Out[(long)(i0 + g * 8 + r) * TD + n0 + l] = acc[r];
}

// ---------------- host-side orchestration ----------------

static inline int gridFor(long n, int bs) { return (int)((n + bs - 1) / bs); }

extern "C" void kernel_launch(void* const* d_in, const int* in_sizes, int n_in,
                              void* d_out, int out_size, void* d_ws, size_t ws_size,
                              hipStream_t stream) {
  const float* x    = (const float*)d_in[0];
  const float* pe   = (const float*)d_in[1];
  const int*   ei   = (const int*)d_in[2];
  const float* W1l  = (const float*)d_in[3];
  const float* b1   = (const float*)d_in[4];
  const float* W1r  = (const float*)d_in[5];
  const float* W2l  = (const float*)d_in[6];
  const float* b2   = (const float*)d_in[7];
  const float* W2r  = (const float*)d_in[8];
  const float* Wa   = (const float*)d_in[9];
  const float* ba   = (const float*)d_in[10];
  const float* Wm   = (const float*)d_in[11];
  const float* bias = (const float*)d_in[12];
  float* out = (float*)d_out;

  const int N  = in_sizes[1] / POSD;     // 4096
  const int E  = in_sizes[2] / 2;        // 131072
  const int H1 = in_sizes[4];            // 68
  const int H2 = in_sizes[7];            // 34

  // carve workspace (floats)
  float* w = (float*)d_ws;
  float* xc      = w;  w += (long)N * TIN;
  float* agg1    = w;  w += (long)N * TIN;
  float* cnt_tgt = w;  w += N;
  float* deg_src = w;  w += N;
  float* dinv    = w;  w += N;
  float* h1buf   = w;  w += (long)N * H1;
  float* agg2    = w;  w += (long)N * H1;
  float* h2buf   = w;  w += (long)N * H2;
  float* cs      = w;  w += 2L * N;
  float* Ox      = w;  w += (long)N * TD;
  float* Wx      = w;  w += (long)N * TD;
  float* hbuf    = w;  w += (long)N * TD;
  float* hsync   = w;  w += (long)N * TD;
  float* term    = w;  w += (long)N * TD;
  float* tmp     = w;  w += (long)N * TD;
  float* result  = w;  w += (long)N * TD;
  (void)ws_size; (void)n_in; (void)out_size;

  const int BS = 256;

  // --- bundle-map learner: SAGE1 -> SAGE2 -> angle head ---
  zero_f32<<<gridFor((long)N * TIN, BS), BS, 0, stream>>>(agg1, N * TIN);
  zero_f32<<<gridFor(N, BS), BS, 0, stream>>>(cnt_tgt, N);
  zero_f32<<<gridFor(N, BS), BS, 0, stream>>>(deg_src, N);
  zero_f32<<<gridFor((long)N * H1, BS), BS, 0, stream>>>(agg2, N * H1);

  concat_kernel<<<gridFor((long)N * TIN, BS), BS, 0, stream>>>(x, pe, xc, N);
  count_kernel<<<gridFor(E, BS), BS, 0, stream>>>(ei, cnt_tgt, deg_src, E);
  dinv_kernel<<<gridFor(N, BS), BS, 0, stream>>>(deg_src, dinv, N);

  scatter_add_kernel<<<gridFor((long)E * (TIN / 4), BS), BS, 0, stream>>>(
      ei, xc, agg1, E, TIN, TIN / 4);
  sage_kernel<<<gridFor((long)N * H1, BS), BS, 0, stream>>>(
      xc, agg1, cnt_tgt, W1l, b1, W1r, h1buf, N, TIN, H1);

  scatter_add_kernel<<<gridFor((long)E * (H1 / 4), BS), BS, 0, stream>>>(
      ei, h1buf, agg2, E, H1, H1 / 4);
  sage_kernel<<<gridFor((long)N * H2, BS), BS, 0, stream>>>(
      h1buf, agg2, cnt_tgt, W2l, b2, W2r, h2buf, N, H1, H2);

  angle_kernel<<<gridFor(N, BS), BS, 0, stream>>>(h2buf, Wa, ba, cs, N, H2);

  // --- h = O^T W O x + bias  (GEMM on the matrix core) ---
  rotate_kernel<<<gridFor((long)N * NBUN, BS), BS, 0, stream>>>(
      cs, x, nullptr, Ox, N, /*transpose=*/0, /*relu=*/0);
  gemm_wmma_kernel<<<N / 16, 256, 0, stream>>>(Ox, Wm, Wx, N);
  rotate_kernel<<<gridFor((long)N * NBUN, BS), BS, 0, stream>>>(
      cs, Wx, bias, hbuf, N, /*transpose=*/1, /*relu=*/0);

  // --- flat bundle diffusion: expm(-tL) @ (O h) via sparse Taylor ---
  rotate_kernel<<<gridFor((long)N * NBUN, BS), BS, 0, stream>>>(
      cs, hbuf, nullptr, hsync, N, /*transpose=*/0, /*relu=*/0);
  diff_init_kernel<<<gridFor((long)N * TD, BS), BS, 0, stream>>>(
      hsync, term, result, N * TD);

  for (int k = 1; k <= MAXDEG; ++k) {
    zero_f32<<<gridFor((long)N * TD, BS), BS, 0, stream>>>(tmp, N * TD);
    edge_apply_kernel<<<gridFor((long)E * 32, BS), BS, 0, stream>>>(
        ei, dinv, term, tmp, E);
    diff_update_kernel<<<gridFor((long)N * TD, BS), BS, 0, stream>>>(
        term, tmp, result, -TDIFF / (float)k, N * TD);
  }

  // --- out = relu(O^T @ diffused) ---
  rotate_kernel<<<gridFor((long)N * NBUN, BS), BS, 0, stream>>>(
      cs, result, nullptr, out, N, /*transpose=*/1, /*relu=*/1);
}
